// RWKV7Attention_65979287601736
// MI455X (gfx1250) — matrix operand order repaired and checked by reading, not compiled, not run on hardware
//
#include <hip/hip_runtime.h>
#include <hip/hip_bf16.h>
#include <math.h>

typedef __attribute__((ext_vector_type(16))) _Float16 v16h;
typedef __attribute__((ext_vector_type(8)))  float    v8f;

#define W_SCALE_C (-0.6065306597126334f) /* -exp(-0.5) */

__device__ __forceinline__ float sigmoidf_(float x) { return 1.0f / (1.0f + expf(-x)); }

__device__ __forceinline__ unsigned pack2h(float a, float b) {
    _Float16 ha = (_Float16)a, hb = (_Float16)b;
    unsigned short ua = __builtin_bit_cast(unsigned short, ha);
    unsigned short ub = __builtin_bit_cast(unsigned short, hb);
    return (unsigned)ua | ((unsigned)ub << 16);
}

// ---------------------------------------------------------------------------
// f16-WMMA GEMM, template-specialized:  C[M,N] = epi( A'[M,K] @ Bw[K,N] + bias )
//   AMODE 0: A' = A (row-major, lda)
//   AMODE 1: A' = token-shift mix of hidden: h + (h[t-1]-h)*xx[k]  (lda = D)
//   AMODE 2: A' = tanh(A)
//   EPI 0: none; 1: sigmoid(x+bias); 2: W_SCALE*sigmoid(x+bias)
// Block tile 128x128, K-step 32. 256 threads = 8 waves (4 over M, 2 over N);
// each wave: 2 A-frags x 4 B-frags -> 8 v_wmma_f32_16x16x32_f16 per K-step.
// ---------------------------------------------------------------------------
template <int AMODE, int EPI>
__global__ __launch_bounds__(256)
void gemm16_kernel(const float* __restrict__ A, const float* __restrict__ Bw,
                   float* __restrict__ C, int M, int N, int K, int lda,
                   const float* __restrict__ xx, int Tlen,
                   const float* __restrict__ bias)
{
    // Fragment-order staging (packed f16 pairs): lane reads 32 contiguous bytes.
    __shared__ __align__(32) unsigned Af[8][32][8]; // [m-subtile][lane][pair]
    __shared__ __align__(32) unsigned Bf[8][32][8]; // [n-subtile][lane][pair]

    const int tid  = threadIdx.x;
    const int wv   = tid >> 5;
    const int lane = tid & 31;
    const int m0 = blockIdx.y * 128;
    const int n0 = blockIdx.x * 128;
    const int wm = (wv & 3) * 2;   // wave's first m-subtile (owns 2)
    const int wn = (wv >> 2) * 4;  // wave's first n-subtile (owns 4)

    v8f acc[2][4];
    #pragma unroll
    for (int i = 0; i < 2; ++i)
        #pragma unroll
        for (int n = 0; n < 4; ++n) acc[i][n] = (v8f){};

    for (int k0 = 0; k0 < K; k0 += 32) {
        if (k0 + 32 < K) { // prefetch next K-tile (global_prefetch_b8)
            __builtin_prefetch(&Bw[(size_t)(k0 + 32) * N + n0 + (tid & 127)], 0, 0);
            __builtin_prefetch(&A[(size_t)(m0 + (tid >> 1)) * lda + k0 + 32], 0, 0);
        }
        // ---- stage A tile (128x32) as packed f16 pairs ----
        #pragma unroll
        for (int i = 0; i < 8; ++i) {
            int p   = i * 256 + tid;        // pair index, 0..2047
            int row = p >> 4;               // 0..127
            int kk  = (p & 15) * 2;         // even k within tile
            int m = m0 + row, kg = k0 + kk; // K is even -> one guard for both
            float v0 = 0.0f, v1 = 0.0f;
            if (m < M && kg < K) {
                if (AMODE == 1) {
                    float2 hc = *(const float2*)&A[(size_t)m * lda + kg];
                    float2 hs = make_float2(0.0f, 0.0f);
                    if ((m % Tlen) != 0) hs = *(const float2*)&A[(size_t)(m - 1) * lda + kg];
                    float2 xv = *(const float2*)&xx[kg];
                    v0 = hc.x + (hs.x - hc.x) * xv.x;
                    v1 = hc.y + (hs.y - hc.y) * xv.y;
                } else {
                    float2 av = *(const float2*)&A[(size_t)m * lda + kg];
                    v0 = av.x; v1 = av.y;
                    if (AMODE == 2) { v0 = tanhf(v0); v1 = tanhf(v1); }
                }
            }
            // 16-bit A 16x32 layout: l = (row&15)|(kk&8?16:0); e = (kk&7)+(kk&16?8:0)
            int l = (row & 15) | ((kk & 8) ? 16 : 0);
            int e = (kk & 7) + ((kk & 16) ? 8 : 0);
            Af[row >> 4][l][e >> 1] = pack2h(v0, v1);
        }
        // ---- stage B tile (32x128) as packed f16 pairs ----
        #pragma unroll
        for (int i = 0; i < 8; ++i) {
            int p  = i * 256 + tid;         // pair index, 0..2047
            int n  = p & 127;               // fast -> coalesced over n
            int kk = (p >> 7) * 2;
            int kg = k0 + kk, ng = n0 + n;
            float v0 = 0.0f, v1 = 0.0f;
            if (ng < N && kg < K) {
                v0 = Bw[(size_t)kg * N + ng];
                v1 = Bw[(size_t)(kg + 1) * N + ng];
            }
            // 16-bit B 32x16 layout: l = (n&15)|(kk&8?16:0); e = (kk&7)+(kk&16?8:0)
            int l = (n & 15) | ((kk & 8) ? 16 : 0);
            int e = (kk & 7) + ((kk & 16) ? 8 : 0);
            Bf[n >> 4][l][e >> 1] = pack2h(v0, v1);
        }
        __syncthreads();

        v16h a0 = *(const v16h*)(&Af[wm + 0][lane][0]);
        v16h a1 = *(const v16h*)(&Af[wm + 1][lane][0]);
        #pragma unroll
        for (int n = 0; n < 4; ++n) {
            v16h b = *(const v16h*)(&Bf[wn + n][lane][0]);
            acc[0][n] = __builtin_amdgcn_wmma_f32_16x16x32_f16(false, a0, false, b, (short)0, acc[0][n], false, false);
            acc[1][n] = __builtin_amdgcn_wmma_f32_16x16x32_f16(false, a1, false, b, (short)0, acc[1][n], false, false);
        }
        __syncthreads();
    }

    #pragma unroll
    for (int n = 0; n < 4; ++n) {
        int col = n0 + (wn + n) * 16 + (lane & 15);
        if (col >= N) continue;
        float bval = (EPI != 0 && bias != nullptr) ? bias[col] : 0.0f;
        #pragma unroll
        for (int i = 0; i < 2; ++i) {
            #pragma unroll
            for (int j = 0; j < 8; ++j) {
                int m = m0 + (wm + i) * 16 + j + ((lane >= 16) ? 8 : 0);
                if (m >= M) continue;
                float v = acc[i][n][j];
                if (EPI == 1)      v = sigmoidf_(v + bval);
                else if (EPI == 2) v = W_SCALE_C * sigmoidf_(v + bval);
                C[(size_t)m * N + col] = v;
            }
        }
    }
}

// ---------------------------------------------------------------------------
// Prep: per (token m, head h) wave:  kk normalize, k/v update, exp(w), a_in/b_in
// ---------------------------------------------------------------------------
__global__ __launch_bounds__(256)
void prep_kernel(float* __restrict__ Kb, float* __restrict__ Vb, float* __restrict__ Wb,
                 const float* __restrict__ Ab, const float* __restrict__ VS,
                 const float* __restrict__ vfirst, const float* __restrict__ kkw,
                 const float* __restrict__ kaw,
                 float* __restrict__ AIN, float* __restrict__ BIN, int M, int D, int H)
{
    int lane = threadIdx.x & 31;
    int pair = blockIdx.x * 8 + (threadIdx.x >> 5);
    if (pair >= M * H) return;
    int m = pair / H, h = pair % H;
    size_t base = (size_t)m * D + h * 64;

    float k0v[2], kkv[2];
    float ss = 0.0f;
    #pragma unroll
    for (int e = 0; e < 2; ++e) {
        int d = lane + e * 32;
        float k0 = Kb[base + d];
        float kk = k0 * kkw[h * 64 + d];
        k0v[e] = k0; kkv[e] = kk;
        ss += kk * kk;
    }
    for (int msk = 16; msk; msk >>= 1) ss += __shfl_xor(ss, msk, 32);
    float rn = rsqrtf(ss + 1e-12f);

    #pragma unroll
    for (int e = 0; e < 2; ++e) {
        int d = lane + e * 32;
        size_t idx = base + d;
        float kkn = kkv[e] * rn;
        float a   = Ab[idx];
        AIN[idx] = -kkn;
        BIN[idx] = kkn * a;
        Kb[idx]  = k0v[e] + (k0v[e] * (a - 1.0f)) * kaw[h * 64 + d];
        float v0 = Vb[idx];
        Vb[idx]  = v0 + VS[idx] * (vfirst[idx] - v0);
        Wb[idx]  = expf(Wb[idx]);
    }
}

// ---------------------------------------------------------------------------
// Delta-rule scan: one block per (batch, head); 64x64 fp32 state in registers,
// thread owns 16 state columns of one row; 4-lane shuffle reductions.
// ---------------------------------------------------------------------------
__global__ __launch_bounds__(256)
void scan_kernel(const float* __restrict__ R, const float* __restrict__ Wexp,
                 const float* __restrict__ Kv, const float* __restrict__ Vv,
                 const float* __restrict__ AIN, const float* __restrict__ BIN,
                 float* __restrict__ O, int Tlen, int D, int H)
{
    int tid = threadIdx.x;
    int bh = blockIdx.x;
    int bb = bh / H, h = bh % H;
    int vr = tid >> 2;           // state row (value dim), 0..63
    int cg = tid & 3;            // column group
    int c0 = cg * 16;

    float S[16];
    #pragma unroll
    for (int j = 0; j < 16; ++j) S[j] = 0.0f;

    __shared__ float sh[384];    // r | w | k | v | a | b, 64 each

    for (int t = 0; t < Tlen; ++t) {
        size_t rowoff = ((size_t)(bb * Tlen + t)) * D + h * 64;
        for (int i = tid; i < 384; i += 256) {
            int which = i >> 6, d = i & 63;
            const float* src = (which == 0) ? R : (which == 1) ? Wexp : (which == 2) ? Kv
                             : (which == 3) ? Vv : (which == 4) ? AIN : BIN;
            sh[i] = src[rowoff + d];
        }
        __syncthreads();

        float sa = 0.0f;
        #pragma unroll
        for (int j = 0; j < 16; ++j) sa += S[j] * sh[256 + c0 + j];  // S @ a_in
        sa += __shfl_xor(sa, 1, 32);
        sa += __shfl_xor(sa, 2, 32);

        float vrow = sh[192 + vr];
        float op = 0.0f;
        #pragma unroll
        for (int j = 0; j < 16; ++j) {
            S[j] = S[j] * sh[64 + c0 + j] + sa * sh[320 + c0 + j] + vrow * sh[128 + c0 + j];
            op += S[j] * sh[c0 + j];                                 // S @ r
        }
        op += __shfl_xor(op, 1, 32);
        op += __shfl_xor(op, 2, 32);
        if (cg == 0) O[rowoff + vr] = op;
        __syncthreads();
    }
}

// ---------------------------------------------------------------------------
// Post: groupnorm per head + bonus + gate -> Y
// ---------------------------------------------------------------------------
__global__ __launch_bounds__(256)
void post_kernel(const float* __restrict__ O, const float* __restrict__ R,
                 const float* __restrict__ Kb, const float* __restrict__ Vb,
                 const float* __restrict__ Gb, const float* __restrict__ rk,
                 const float* __restrict__ gnw, const float* __restrict__ gnb,
                 float* __restrict__ Y, int M, int D, int H)
{
    int lane = threadIdx.x & 31;
    int pair = blockIdx.x * 8 + (threadIdx.x >> 5);
    if (pair >= M * H) return;
    int m = pair / H, h = pair % H;
    size_t base = (size_t)m * D + h * 64;

    float so = 0.0f, so2 = 0.0f, srk = 0.0f;
    float ov[2];
    #pragma unroll
    for (int e = 0; e < 2; ++e) {
        int d = lane + e * 32;
        size_t idx = base + d;
        float o = O[idx]; ov[e] = o;
        so += o; so2 += o * o;
        srk += R[idx] * Kb[idx] * rk[h * 64 + d];
    }
    for (int msk = 16; msk; msk >>= 1) {
        so  += __shfl_xor(so,  msk, 32);
        so2 += __shfl_xor(so2, msk, 32);
        srk += __shfl_xor(srk, msk, 32);
    }
    float mu   = so * (1.0f / 64.0f);
    float var  = so2 * (1.0f / 64.0f) - mu * mu;
    float rstd = rsqrtf(var + 64.0f * 1e-5f);   // eps = head_dim * 1e-5

    #pragma unroll
    for (int e = 0; e < 2; ++e) {
        int d = lane + e * 32;
        size_t idx = base + d;
        float on = (ov[e] - mu) * rstd * gnw[h * 64 + d] + gnb[h * 64 + d];
        Y[idx] = (on + srk * Vb[idx]) * Gb[idx];
    }
}

// ---------------------------------------------------------------------------
template <int AMODE, int EPI>
static void launch_gemm_t(const float* A, const float* Bw, float* C, int M, int N, int K,
                          int lda, const float* xx, int Tlen, const float* bias,
                          hipStream_t s)
{
    dim3 grid((N + 127) / 128, (M + 127) / 128);
    gemm16_kernel<AMODE, EPI><<<grid, 256, 0, s>>>(A, Bw, C, M, N, K, lda, xx, Tlen, bias);
}

extern "C" void kernel_launch(void* const* d_in, const int* in_sizes, int n_in,
                              void* d_out, int out_size, void* d_ws, size_t ws_size,
                              hipStream_t stream)
{
    const int Bsz = 8, Tlen = 2048, D = 1024, H = 16;
    const int M = Bsz * Tlen;

    const float* hidden = (const float*)d_in[0];
    const float* vfirst = (const float*)d_in[1];
    const float* x_x    = (const float*)d_in[2];   // [6, D]
    const float* k_k    = (const float*)d_in[3];
    const float* k_a    = (const float*)d_in[4];
    const float* r_k    = (const float*)d_in[5];
    const float* W_r    = (const float*)d_in[6];
    const float* W_k    = (const float*)d_in[7];
    const float* W_v    = (const float*)d_in[8];
    const float* W_o    = (const float*)d_in[9];
    const float* wA     = (const float*)d_in[10];
    const float* wB     = (const float*)d_in[11];
    const float* wb     = (const float*)d_in[12];
    const float* vA     = (const float*)d_in[13];
    const float* vB     = (const float*)d_in[14];
    const float* vb     = (const float*)d_in[15];
    const float* aA     = (const float*)d_in[16];
    const float* aB     = (const float*)d_in[17];
    const float* ab     = (const float*)d_in[18];
    const float* gA     = (const float*)d_in[19];
    const float* gB     = (const float*)d_in[20];
    const float* gnw    = (const float*)d_in[21];
    const float* gnb    = (const float*)d_in[22];
    float* out = (float*)d_out;

    float* ws = (float*)d_ws;
    const size_t MD = (size_t)M * D;
    float* R   = ws + 0 * MD;
    float* Kb  = ws + 1 * MD;
    float* Vb  = ws + 2 * MD;
    float* Wb  = ws + 3 * MD;
    float* Ab  = ws + 4 * MD;
    float* VS  = ws + 5 * MD;
    float* Gb  = ws + 6 * MD;
    float* AIN = ws + 7 * MD;
    float* BIN = ws + 8 * MD;
    float* Ob  = VS;            // VS dead after prep -> scan output
    float* Yb  = Ab;            // Ab dead after prep -> gated output
    // LoRA stage-1 intermediates live only until stage-2 GEMMs; park in AIN slot.
    float* wl1 = AIN;
    float* vl1 = wl1 + (size_t)M * 64;
    float* al1 = vl1 + (size_t)M * 16;
    float* gl1 = al1 + (size_t)M * 64;
    (void)in_sizes; (void)n_in; (void)out_size; (void)ws_size;

    // --- Stage 1: projections from token-shift-mixed hidden (mix fused in A load)
    launch_gemm_t<1, 0>(hidden, W_r, R,   M, D,   D, D, x_x + 0 * D, Tlen, nullptr, stream);
    launch_gemm_t<1, 0>(hidden, W_k, Kb,  M, D,   D, D, x_x + 2 * D, Tlen, nullptr, stream);
    launch_gemm_t<1, 0>(hidden, W_v, Vb,  M, D,   D, D, x_x + 3 * D, Tlen, nullptr, stream);
    launch_gemm_t<1, 0>(hidden, wA,  wl1, M, 64,  D, D, x_x + 1 * D, Tlen, nullptr, stream);
    launch_gemm_t<1, 0>(hidden, vA,  vl1, M, 16,  D, D, x_x + 3 * D, Tlen, nullptr, stream);
    launch_gemm_t<1, 0>(hidden, aA,  al1, M, 64,  D, D, x_x + 4 * D, Tlen, nullptr, stream);
    launch_gemm_t<1, 0>(hidden, gA,  gl1, M, 128, D, D, x_x + 5 * D, Tlen, nullptr, stream);

    // --- Stage 2: LoRA second stages with fused activations
    launch_gemm_t<2, 2>(wl1, wB, Wb, M, D, 64,  64,  nullptr, Tlen, wb,      stream);
    launch_gemm_t<0, 1>(vl1, vB, VS, M, D, 16,  16,  nullptr, Tlen, vb,      stream);
    launch_gemm_t<0, 1>(al1, aB, Ab, M, D, 64,  64,  nullptr, Tlen, ab,      stream);
    launch_gemm_t<0, 1>(gl1, gB, Gb, M, D, 128, 128, nullptr, Tlen, nullptr, stream);

    // --- Stage 3: elementwise prep (normalize kk, k/v update, exp(w), a_in/b_in)
    prep_kernel<<<(M * H) / 8, 256, 0, stream>>>(Kb, Vb, Wb, Ab, VS, vfirst, k_k, k_a,
                                                 AIN, BIN, M, D, H);

    // --- Stage 4: sequential delta-rule scan, one block per (B,H) chain
    scan_kernel<<<Bsz * H, 256, 0, stream>>>(R, Wb, Kb, Vb, AIN, BIN, Ob, Tlen, D, H);

    // --- Stage 5: groupnorm + bonus + gate
    post_kernel<<<(M * H) / 8, 256, 0, stream>>>(Ob, R, Kb, Vb, Gb, r_k, gnw, gnb,
                                                 Yb, M, D, H);

    // --- Stage 6: output projection
    launch_gemm_t<0, 0>(Yb, W_o, out, M, D, D, D, nullptr, Tlen, nullptr, stream);
}
